// AttentionHead_77017353552430
// MI455X (gfx1250) — matrix operand order repaired and checked by reading, not compiled, not run on hardware
//
#include <hip/hip_runtime.h>
#include <cstdint>

// ---------------------------------------------------------------------------
// Single-head attention x:[4,4096,768], head=64 on gfx1250 (wave32, WMMA).
// f16 WMMA with fp32 accumulation. Block-cooperative DOUBLE-BUFFERED async
// LDS staging (GLOBAL_LOAD_ASYNC_TO_LDS_B128 / ASYNCcnt). Chunk loops stay
// un-unrolled (single body) so accumulators remain pinned; B-fragments are
// preloaded in groups so ds_load -> WMMA retires with partial dscnt waits.
// ---------------------------------------------------------------------------

typedef _Float16 f16;
typedef __attribute__((ext_vector_type(4)))  _Float16 v4h;
typedef __attribute__((ext_vector_type(8)))  _Float16 v8h;
typedef __attribute__((ext_vector_type(16))) _Float16 v16h;
typedef __attribute__((ext_vector_type(8)))  float    v8f;
typedef int v4i __attribute__((vector_size(16)));   // matches builtin pointee

#define HIDDEN 768
#define SEQ    4096
#define HEAD   64
#define BATCH  4
#define NW     (HIDDEN * HEAD)

#if defined(__HIP_DEVICE_COMPILE__) && \
    __has_builtin(__builtin_amdgcn_global_load_async_to_lds_b128)
#define USE_ASYNC_LDS 1
#else
#define USE_ASYNC_LDS 0
#endif

// 16B global -> LDS copy. Async path: GLOBAL_LOAD_ASYNC_TO_LDS_B128.
// Generic pointers to LDS carry the LDS byte offset in their low 32 bits
// (ISA 10.2), so the integer-bridged casts are value-correct.
static __device__ __forceinline__ void stage16(f16* lds_dst, const f16* g_src) {
#if USE_ASYNC_LDS
  __builtin_amdgcn_global_load_async_to_lds_b128(
      (__attribute__((address_space(1))) v4i*)(uintptr_t)g_src,
      (__attribute__((address_space(3))) v4i*)(unsigned)(uintptr_t)lds_dst,
      0, 0);
#else
  *reinterpret_cast<v8h*>(lds_dst) = *reinterpret_cast<const v8h*>(g_src);
#endif
}

static __device__ __forceinline__ void async_wait() {
#if USE_ASYNC_LDS
  asm volatile("s_wait_asynccnt 0" ::: "memory");
#endif
}

static __device__ __forceinline__ v8f wmma16x16x32(v16h a, v16h b, v8f c) {
  return __builtin_amdgcn_wmma_f32_16x16x32_f16(false, a, false, b, (short)0, c,
                                                false, false);
}

// A-matrix 16x32 f16 fragment (two contiguous 16B runs per lane).
static __device__ __forceinline__ v16h load_A16x32(const f16* base, int stride,
                                                   int lane, int kbase) {
  const int row = lane & 15;
  const int hf  = (lane >> 4) & 1;
  const f16* p = base + row * stride + kbase + hf * 8;
  v8h lo = *reinterpret_cast<const v8h*>(p);
  v8h hi = *reinterpret_cast<const v8h*>(p + 16);
  v16h a;
#pragma unroll
  for (int i = 0; i < 8; ++i) { a[i] = lo[i]; a[i + 8] = hi[i]; }
  return a;
}

// B-matrix 32x16 f16 fragment (two contiguous 16B runs per lane).
static __device__ __forceinline__ v16h load_B32x16(const f16* base, int stride,
                                                   int lane, int kbase) {
  const int col = lane & 15;
  const int hf  = (lane >> 4) & 1;
  const f16* p = base + col * stride + kbase + hf * 8;
  v8h lo = *reinterpret_cast<const v8h*>(p);
  v8h hi = *reinterpret_cast<const v8h*>(p + 16);
  v16h b;
#pragma unroll
  for (int i = 0; i < 8; ++i) { b[i] = lo[i]; b[i + 8] = hi[i]; }
  return b;
}

static __device__ __forceinline__ float rowmax16(float v) {
#pragma unroll
  for (int m = 8; m; m >>= 1) v = fmaxf(v, __shfl_xor(v, m, 16));
  return v;
}
static __device__ __forceinline__ float rowsum16(float v) {
#pragma unroll
  for (int m = 8; m; m >>= 1) v += __shfl_xor(v, m, 16);
  return v;
}

// ---------------------------------------------------------------------------
// Kernel 1: x fp32 -> f16 (HBM streaming, 4-wide).
// ---------------------------------------------------------------------------
__global__ void k_cvt_x(const float4* __restrict__ x, v4h* __restrict__ x16,
                        int n4) {
  for (int i = blockIdx.x * blockDim.x + threadIdx.x; i < n4;
       i += gridDim.x * blockDim.x) {
    float4 f = x[i];
    v4h h; h[0] = (f16)f.x; h[1] = (f16)f.y; h[2] = (f16)f.z; h[3] = (f16)f.w;
    x16[i] = h;
  }
}

// ---------------------------------------------------------------------------
// Kernel 2: weights fp32 -> f16 transposed.
//   WT (q,k,v consecutive): [3][HEAD][HIDDEN];  WoT: [HIDDEN][HEAD]
// ---------------------------------------------------------------------------
__global__ void k_cvt_w(const float* __restrict__ Wq, const float* __restrict__ Wk,
                        const float* __restrict__ Wv, const float* __restrict__ Wo,
                        f16* __restrict__ WT, f16* __restrict__ WoT) {
  for (int i = blockIdx.x * blockDim.x + threadIdx.x; i < NW;
       i += gridDim.x * blockDim.x) {
    int h = i / HEAD, d = i - h * HEAD;        // W[h][d]
    WT[0 * NW + d * HIDDEN + h] = (f16)Wq[i];
    WT[1 * NW + d * HIDDEN + h] = (f16)Wk[i];
    WT[2 * NW + d * HIDDEN + h] = (f16)Wv[i];
    int d2 = i / HIDDEN, h2 = i - d2 * HIDDEN; // Wo[d2][h2]
    WoT[h2 * HEAD + d2] = (f16)Wo[i];
  }
}

// ---------------------------------------------------------------------------
// Kernel 3: QKV projection, split per matrix (grid = 3 x 256 blocks).
// 4 waves/block, one 16-row tile each. Double-buffered 4KB weight slice
// [64][32] per k-step staged async into LDS; one barrier per k-step.
// ---------------------------------------------------------------------------
__global__ void __launch_bounds__(128) k_qkv(
    const f16* __restrict__ X16, const f16* __restrict__ WT,
    const float* __restrict__ bq, const float* __restrict__ bk,
    const float* __restrict__ bv, f16* __restrict__ Q, f16* __restrict__ K,
    f16* __restrict__ VT) {
  __shared__ __align__(16) f16 wbuf[2][HEAD * 32];   // 2 x 4KB
  const int tid  = threadIdx.x;
  const int lane = tid & 31;
  const int wid  = tid >> 5;
  const int mat  = blockIdx.x >> 8;            // 0=Q 1=K 2=V
  const int tile = (blockIdx.x & 255) * 4 + wid;
  const int b    = tile >> 8;                  // same batch per block
  const int row0 = (tile & 255) << 4;

  const f16* Arow = X16 + (size_t)(b * SEQ + row0) * HIDDEN;
  const f16* Wmat = WT + (size_t)mat * NW;

  auto stage_w = [&](int par, int kc) {
#pragma unroll
    for (int i = 0; i < 2; ++i) {
      const int p = i * 128 + tid;             // 256 pieces of 16B
      const int row = p >> 2, part = p & 3;    // row in [0,64)
      stage16(&wbuf[par][row * 32 + part * 8],
              Wmat + row * HIDDEN + kc + part * 8);
    }
  };

  v8f acc[4];
#pragma unroll
  for (int t = 0; t < 4; ++t) acc[t] = (v8f)0.f;

  stage_w(0, 0);
  int par = 0;
#pragma clang loop unroll(disable)
  for (int kc = 0; kc < HIDDEN; kc += 32) {
    v16h a = load_A16x32(Arow, HIDDEN, lane, kc);  // issue before barrier wait
    async_wait();
    __syncthreads();
    if (kc + 32 < HIDDEN) stage_w(par ^ 1, kc + 32);  // DMA overlaps compute
    // Preload all 4 B-fragments, then run the WMMA chain (partial waits).
    v16h bf[4];
#pragma unroll
    for (int t = 0; t < 4; ++t)
      bf[t] = load_B32x16(&wbuf[par][t * 16 * 32], 32, lane, 0);
#pragma unroll
    for (int t = 0; t < 4; ++t) acc[t] = wmma16x16x32(a, bf[t], acc[t]);
    par ^= 1;
  }

  const int hf  = (lane >> 4) & 1;
  const int col = lane & 15;
  const float* bias = (mat == 0) ? bq : (mat == 1) ? bk : bv;
  if (mat < 2) {
    f16* dst = (mat == 0) ? Q : K;
    const float scale = (mat == 0) ? 0.125f : 1.0f;   // fold 1/sqrt(64) into Q
#pragma unroll
    for (int t = 0; t < 4; ++t) {
      const int d = t * 16 + col;
      const float bi = bias[d];
#pragma unroll
      for (int j = 0; j < 8; ++j) {
        const int row = row0 + j + hf * 8;
        dst[(size_t)(b * SEQ + row) * HEAD + d] = (f16)((acc[t][j] + bi) * scale);
      }
    }
  } else {
#pragma unroll
    for (int t = 0; t < 4; ++t) {
      const int d = t * 16 + col;
      const float bi = bias[d];
#pragma unroll
      for (int j = 0; j < 8; ++j) {
        const int row = row0 + j + hf * 8;
        VT[(size_t)(b * HEAD + d) * SEQ + row] = (f16)(acc[t][j] + bi);
      }
    }
  }
}

// ---------------------------------------------------------------------------
// Kernel 4: flash attention + fused output projection. 4 waves/block (same
// batch), 256 blocks. Double-buffered 8KB K/V chunk staging (async LDS),
// one barrier per chunk. Each wave owns 16 query rows; P and ctx transpose
// C-layout -> A-layout via a per-wave LDS tile.
// ---------------------------------------------------------------------------
__global__ void __launch_bounds__(128) k_attn(
    const f16* __restrict__ Q, const f16* __restrict__ K,
    const f16* __restrict__ VT, const f16* __restrict__ WoT,
    const float* __restrict__ bo, float* __restrict__ out) {
  __shared__ __align__(16) f16 kbuf[2][32 * HEAD];     // 2 x 4KB [key][dim]
  __shared__ __align__(16) f16 vbuf[2][HEAD * 32];     // 2 x 4KB [dim][key]
  __shared__ __align__(16) f16 pbuf[4 * 16 * HEAD];    // 8KB per-wave tiles
  const int tid  = threadIdx.x;
  const int lane = tid & 31;
  const int wid  = tid >> 5;
  f16* myLds = &pbuf[wid * 16 * HEAD];

  const int tile = blockIdx.x * 4 + wid;
  const int b    = tile >> 8;                  // same batch per block
  const int row0 = (tile & 255) << 4;
  const int hf   = (lane >> 4) & 1;
  const int col  = lane & 15;

  const f16* Kbat  = K  + (size_t)b * SEQ * HEAD;
  const f16* VTbat = VT + (size_t)b * HEAD * SEQ;

  auto stage_kv = [&](int par, int kv0) {
#pragma unroll
    for (int i = 0; i < 2; ++i) {
      const int p = i * 128 + tid;             // 256 pieces of 16B each buffer
      stage16(&kbuf[par][(p >> 3) * HEAD + (p & 7) * 8],
              Kbat + (size_t)(kv0 + (p >> 3)) * HEAD + (p & 7) * 8);
      stage16(&vbuf[par][(p >> 2) * 32 + (p & 3) * 8],
              VTbat + (size_t)(p >> 2) * SEQ + kv0 + (p & 3) * 8);
    }
  };

  const f16* Qrow = Q + (size_t)(b * SEQ + row0) * HEAD;
  v16h aQ0 = load_A16x32(Qrow, HEAD, lane, 0);
  v16h aQ1 = load_A16x32(Qrow, HEAD, lane, 32);

  v8f acc[4];
#pragma unroll
  for (int t = 0; t < 4; ++t) acc[t] = (v8f)0.f;
  float mrow[8], lrow[8];
#pragma unroll
  for (int j = 0; j < 8; ++j) { mrow[j] = -3.0e38f; lrow[j] = 0.f; }

  stage_kv(0, 0);
  int par = 0;
#pragma clang loop unroll(disable)
  for (int kv0 = 0; kv0 < SEQ; kv0 += 32) {
    async_wait();
    __syncthreads();
    if (kv0 + 32 < SEQ) stage_kv(par ^ 1, kv0 + 32);  // DMA overlaps compute

    // ---- scores: preload 4 K B-fragments, then 4 WMMAs ----
    v8f s[2];
    {
      v16h bk[4];
#pragma unroll
      for (int t = 0; t < 2; ++t) {
        bk[t * 2]     = load_B32x16(&kbuf[par][t * 16 * HEAD], HEAD, lane, 0);
        bk[t * 2 + 1] = load_B32x16(&kbuf[par][t * 16 * HEAD], HEAD, lane, 32);
      }
#pragma unroll
      for (int t = 0; t < 2; ++t) {
        v8f c = (v8f)0.f;
        c = wmma16x16x32(aQ0, bk[t * 2],     c);
        c = wmma16x16x32(aQ1, bk[t * 2 + 1], c);
        s[t] = c;
      }
    }
    // ---- online softmax; write P (f16) to per-wave LDS tile ----
#pragma unroll
    for (int j = 0; j < 8; ++j) {
      float rm = rowmax16(fmaxf(s[0][j], s[1][j]));
      float mn = fmaxf(mrow[j], rm);
      float corr = __expf(mrow[j] - mn);
      mrow[j] = mn;
      float p0 = __expf(s[0][j] - mn);
      float p1 = __expf(s[1][j] - mn);
      lrow[j] = lrow[j] * corr + rowsum16(p0 + p1);
#pragma unroll
      for (int t = 0; t < 4; ++t) acc[t][j] *= corr;
      const int row = j + hf * 8;
      myLds[row * 32 + col]      = (f16)p0;
      myLds[row * 32 + 16 + col] = (f16)p1;
    }
    asm volatile("s_wait_dscnt 0" ::: "memory");
    // ---- acc += P(16x32) x V(32x16): preload P and 4 V B-fragments ----
    {
      v16h aP = load_A16x32(myLds, 32, lane, 0);
      v16h bV[4];
#pragma unroll
      for (int t = 0; t < 4; ++t)
        bV[t] = load_B32x16(&vbuf[par][t * 16 * 32], 32, lane, 0);
#pragma unroll
      for (int t = 0; t < 4; ++t) acc[t] = wmma16x16x32(aP, bV[t], acc[t]);
    }
    par ^= 1;
  }

  // ---- normalize; stage ctx (16x64 f16) in per-wave LDS (A-layout source) --
#pragma unroll
  for (int j = 0; j < 8; ++j) {
    const float inv = 1.0f / lrow[j];
    const int row = j + hf * 8;
#pragma unroll
    for (int t = 0; t < 4; ++t)
      myLds[row * HEAD + t * 16 + col] = (f16)(acc[t][j] * inv);
  }
  asm volatile("s_wait_dscnt 0" ::: "memory");
  v16h aC0 = load_A16x32(myLds, HEAD, lane, 0);
  v16h aC1 = load_A16x32(myLds, HEAD, lane, 32);

  // ---- output projection: 48 tiles of 16 hidden cols; WoT tiles double-
  //      buffered through the (now free) kbuf halves ----
  auto stage_wo = [&](int par2, int n0) {
    // 16 rows x 8x16B pieces (2KB); one piece per thread
    stage16(&kbuf[par2][(tid >> 3) * HEAD + (tid & 7) * 8],
            WoT + (size_t)(n0 + (tid >> 3)) * HEAD + (tid & 7) * 8);
  };

  __syncthreads();        // all waves done with flash-phase kbuf reads
  stage_wo(0, 0);
  par = 0;
#pragma clang loop unroll(disable)
  for (int n0 = 0; n0 < HIDDEN; n0 += 16) {
    async_wait();
    __syncthreads();
    if (n0 + 16 < HIDDEN) stage_wo(par ^ 1, n0 + 16);

    v16h bw0 = load_B32x16(kbuf[par], HEAD, lane, 0);
    v16h bw1 = load_B32x16(kbuf[par], HEAD, lane, 32);
    v8f c = (v8f)0.f;
    c = wmma16x16x32(aC0, bw0, c);
    c = wmma16x16x32(aC1, bw1, c);
    const float bias = bo[n0 + col];
#pragma unroll
    for (int j = 0; j < 8; ++j) {
      const int row = row0 + j + hf * 8;
      out[(size_t)(b * SEQ + row) * HIDDEN + n0 + col] = c[j] + bias;
    }
    par ^= 1;
  }
}

// ---------------------------------------------------------------------------
// Host side
// ---------------------------------------------------------------------------
extern "C" void kernel_launch(void* const* d_in, const int* in_sizes, int n_in,
                              void* d_out, int out_size, void* d_ws, size_t ws_size,
                              hipStream_t stream) {
  (void)in_sizes; (void)n_in; (void)out_size; (void)ws_size;
  const float* x  = (const float*)d_in[0];
  const float* Wq = (const float*)d_in[1];
  const float* bq = (const float*)d_in[2];
  const float* Wk = (const float*)d_in[3];
  const float* bk = (const float*)d_in[4];
  const float* Wv = (const float*)d_in[5];
  const float* bv = (const float*)d_in[6];
  const float* Wo = (const float*)d_in[7];
  const float* bo = (const float*)d_in[8];
  float* out = (float*)d_out;

  char* ws = (char*)d_ws;
  const size_t nX = (size_t)BATCH * SEQ * HIDDEN;
  const size_t nQ = (size_t)BATCH * SEQ * HEAD;
  size_t off = 0;
  f16* X16 = (f16*)(ws + off); off += nX * sizeof(f16);
  f16* Qb  = (f16*)(ws + off); off += nQ * sizeof(f16);
  f16* Kb  = (f16*)(ws + off); off += nQ * sizeof(f16);
  f16* VTb = (f16*)(ws + off); off += nQ * sizeof(f16);
  f16* WT  = (f16*)(ws + off); off += 3 * (size_t)NW * sizeof(f16);
  f16* WoT = (f16*)(ws + off); off += (size_t)NW * sizeof(f16);

  hipLaunchKernelGGL(k_cvt_x, dim3(3072), dim3(256), 0, stream,
                     (const float4*)x, (v4h*)X16, (int)(nX / 4));
  hipLaunchKernelGGL(k_cvt_w, dim3(96), dim3(256), 0, stream,
                     Wq, Wk, Wv, Wo, WT, WoT);
  hipLaunchKernelGGL(k_qkv, dim3(3 * 256), dim3(128), 0, stream,
                     X16, WT, bq, bk, bv, Qb, Kb, VTb);
  hipLaunchKernelGGL(k_attn, dim3(256), dim3(128), 0, stream,
                     Qb, Kb, VTb, WoT, bo, out);
}
